// Qwen3MoeDecoderLayer_63952063038003
// MI455X (gfx1250) — matrix operand order repaired and checked by reading, not compiled, not run on hardware
//
#include <hip/hip_runtime.h>

// ---------------- problem constants (B=1) ----------------
constexpr int S   = 2048;
constexpr int H   = 2048;
constexpr int HQ  = 32;
constexpr int HKV = 4;
constexpr int D   = 128;
constexpr int HQD = HQ * D;    // 4096
constexpr int KVD = HKV * D;   // 512
constexpr int E   = 32;
constexpr int TK  = 4;
constexpr int I   = 768;
constexpr int PADROWS = S * TK + E * 16;  // 8704 (16-row padded expert groups)
constexpr int MAXBLK  = PADROWS / 16;     // 544
constexpr float EPS = 1e-6f;

typedef __bf16 bf16;
typedef __attribute__((ext_vector_type(16))) __bf16 v16bf;
typedef __attribute__((ext_vector_type(8)))  __bf16 v8bf;
typedef __attribute__((ext_vector_type(8)))  float  v8f;

__device__ __forceinline__ float bf2f(bf16 b) {
  unsigned short s = __builtin_bit_cast(unsigned short, b);
  unsigned u = ((unsigned)s) << 16;
  return __builtin_bit_cast(float, u);
}
__device__ __forceinline__ bf16 f2bf(float f) {
  unsigned u = __builtin_bit_cast(unsigned, f);
  u += 0x7fffu + ((u >> 16) & 1u);   // round to nearest even
  unsigned short s = (unsigned short)(u >> 16);
  return __builtin_bit_cast(bf16, s);
}

__device__ __forceinline__ v8f wmma_bf16(v16bf a, v16bf b, v8f c) {
  // D = A(16x32) * B(32x16) + C, f32 accumulate
  return __builtin_amdgcn_wmma_f32_16x16x32_bf16(false, a, false, b, (short)0, c, false, false);
}

// LDS byte address of a __shared__ pointer (low 32 bits of the flat address
// are the LDS offset per the aperture mapping).
__device__ __forceinline__ unsigned lds_addr(const void* p) {
  return (unsigned)(unsigned long long)p;
}
// CDNA5 async copy: 16B global -> LDS per lane, tracked by ASYNCcnt.
__device__ __forceinline__ void async_b128(unsigned lds_off, const void* gaddr) {
  asm volatile("global_load_async_to_lds_b128 %0, %1, off"
               :: "v"(lds_off), "v"(gaddr) : "memory");
}
__device__ __forceinline__ void wait_async_le4() {
  asm volatile("s_wait_asynccnt 0x4" ::: "memory");
}
__device__ __forceinline__ void wait_async_le0() {
  asm volatile("s_wait_asynccnt 0x0" ::: "memory");
}

// A fragment: 16x32 bf16 tile, row-major, leading dim lda (elements).
// lane L: row = L&15, holds K = kb..kb+7 and kb+16..kb+23 with kb = (L>=16)*8.
__device__ __forceinline__ v16bf load_a_frag(const bf16* A, int lda) {
  int lane = threadIdx.x & 31;
  const bf16* p = A + (size_t)(lane & 15) * lda + ((lane >> 4) * 8);
  v8bf lo = *(const v8bf*)p;
  v8bf hi = *(const v8bf*)(p + 16);
  v16bf f;
#pragma unroll
  for (int i = 0; i < 8; ++i) { f[i] = lo[i]; f[i + 8] = hi[i]; }
  return f;
}

// B fragment from B^T layout [N x K]: lane L = column L&15, holds K = kb..kb+7 and
// kb+16..kb+23 with kb = (L>=16)*8 (matches A fragment order).
__device__ __forceinline__ v16bf load_bt_frag(const bf16* Bt, int ldb) {
  int lane = threadIdx.x & 31;
  const bf16* p = Bt + (size_t)(lane & 15) * ldb + ((lane >> 4) * 8);
  v8bf lo = *(const v8bf*)p;
  v8bf hi = *(const v8bf*)(p + 16);
  v16bf f;
#pragma unroll
  for (int i = 0; i < 8; ++i) { f[i] = lo[i]; f[i + 8] = hi[i]; }
  return f;
}

// ---------------- weight convert + transpose (fp32 [K][N] -> bf16 [N][K]) ----
__global__ void k_transpose_bf16(const float* __restrict__ in, bf16* __restrict__ out,
                                 int K, int N) {
  size_t base = (size_t)blockIdx.z * K * N;
  size_t gid  = (size_t)blockIdx.x * blockDim.x + threadIdx.x;
  size_t total = (size_t)K * N;
  if (gid >= total) return;
  int n = (int)(gid / K);
  int k = (int)(gid % K);
  out[base + gid] = f2bf(in[base + (size_t)k * N + n]);
}

// ---------------- RMSNorm row kernel (fp32 in, bf16 out) --------------------
__global__ void k_rmsnorm_bf16(const float* __restrict__ x, const float* __restrict__ w,
                               bf16* __restrict__ out, int ncols) {
  __shared__ float red[256];
  int row = blockIdx.x, tid = threadIdx.x;
  const float* xr = x + (size_t)row * ncols;
  float part = 0.f;
  for (int c = tid; c < ncols; c += 256) { float v = xr[c]; part += v * v; }
  red[tid] = part; __syncthreads();
  for (int s2 = 128; s2 > 0; s2 >>= 1) { if (tid < s2) red[tid] += red[tid + s2]; __syncthreads(); }
  float rstd = rsqrtf(red[0] / ncols + EPS);
  bf16* orow = out + (size_t)row * ncols;
  for (int c = tid; c < ncols; c += 256) orow[c] = f2bf(xr[c] * rstd * w[c]);
}

// ---------------- dense GEMM: C[M,N] = A[M,K](bf16) @ Bt[N,K](bf16) ---------
// One wave computes a 32x64 block (2x4 WMMA tiles). The 32-row A strip is
// staged through a double-buffered LDS chunk with global_load_async_to_lds,
// overlapping the next chunk's DMA with the current chunk's 8 WMMAs.
__global__ void k_gemm_bt(const bf16* __restrict__ A, const bf16* __restrict__ Bt,
                          float* __restrict__ C, int N, int Kd) {
  __shared__ bf16 abuf[2 * 32 * 32];     // 2 x (32 rows x 32 k) = 4KB
  int nt0 = blockIdx.x * 4;              // first of 4 n-tiles
  int mt0 = blockIdx.y * 2;              // first of 2 m-tiles
  int lane = threadIdx.x & 31;
  const bf16* Ab = A + (size_t)mt0 * 16 * Kd;    // 32 rows
  const bf16* Bb = Bt + (size_t)nt0 * 16 * Kd;   // 64 cols (4 tiles)
  int srow = lane >> 2;                  // staging: 8 rows per async issue
  int scb  = (lane & 3) * 16;            // 4 lanes x 16B cover one 64B row chunk
  unsigned lbase = lds_addr(abuf);

  v8f acc[2][4];
#pragma unroll
  for (int m = 0; m < 2; ++m)
#pragma unroll
    for (int n = 0; n < 4; ++n)
#pragma unroll
      for (int i = 0; i < 8; ++i) acc[m][n][i] = 0.f;

  int nkc = Kd >> 5;
  // prologue: stage chunk 0 into buffer 0
#pragma unroll
  for (int t = 0; t < 4; ++t) {
    int row = t * 8 + srow;
    async_b128(lbase + row * 64 + scb,
               (const char*)Ab + (size_t)row * Kd * 2 + scb);
  }
  for (int kc = 0; kc < nkc; ++kc) {
    int buf = kc & 1;
    if (kc + 1 < nkc) {                  // stage next chunk, then wait for current
      unsigned dst = lbase + (unsigned)(((kc + 1) & 1) * 2048);
      const char* src = (const char*)Ab + (size_t)(kc + 1) * 64;
#pragma unroll
      for (int t = 0; t < 4; ++t) {
        int row = t * 8 + srow;
        async_b128(dst + row * 64 + scb, src + (size_t)row * Kd * 2 + scb);
      }
      wait_async_le4();                  // in-order: current chunk's 4 copies done
    } else {
      wait_async_le0();
    }
    // A fragments from LDS
    v16bf afr[2];
#pragma unroll
    for (int m = 0; m < 2; ++m) {
      const bf16* p = abuf + buf * 1024 + (m * 16 + (lane & 15)) * 32 + ((lane >> 4) * 8);
      v8bf lo = *(const v8bf*)p;
      v8bf hi = *(const v8bf*)(p + 16);
#pragma unroll
      for (int i = 0; i < 8; ++i) { afr[m][i] = lo[i]; afr[m][i + 8] = hi[i]; }
    }
    // B fragments from global (distinct columns per n-tile)
    v16bf bfr[4];
#pragma unroll
    for (int n = 0; n < 4; ++n)
      bfr[n] = load_bt_frag(Bb + (size_t)n * 16 * Kd + kc * 32, Kd);
#pragma unroll
    for (int m = 0; m < 2; ++m)
#pragma unroll
      for (int n = 0; n < 4; ++n)
        acc[m][n] = wmma_bf16(afr[m], bfr[n], acc[m][n]);
  }
  int col = lane & 15, hi = lane >> 4;
#pragma unroll
  for (int m = 0; m < 2; ++m)
#pragma unroll
    for (int n = 0; n < 4; ++n)
#pragma unroll
      for (int j = 0; j < 8; ++j)
        C[(size_t)((mt0 + m) * 16 + j + hi * 8) * N + (nt0 + n) * 16 + col] = acc[m][n][j];
}

// ---------------- per-head RMSNorm + RoPE (one wave per (s,head)) -----------
__global__ void k_qk_post(const float* __restrict__ qkv, const float* __restrict__ nw,
                          const float* __restrict__ cosb, const float* __restrict__ sinb,
                          bf16* __restrict__ outh, int nheads) {
  int wid  = threadIdx.x >> 5;
  int lane = threadIdx.x & 31;
  int idx  = blockIdx.x * (blockDim.x >> 5) + wid;
  if (idx >= S * nheads) return;
  int s = idx / nheads, hh = idx % nheads;
  int d0 = lane * 4;
  const float* src = qkv + (size_t)s * nheads * D + (size_t)hh * D + d0;
  float x[4]; float ss = 0.f;
#pragma unroll
  for (int i = 0; i < 4; ++i) { x[i] = src[i]; ss += x[i] * x[i]; }
#pragma unroll
  for (int m = 16; m > 0; m >>= 1) ss += __shfl_xor(ss, m, 32);
  float rstd = rsqrtf(ss / D + EPS);
  float xn[4];
#pragma unroll
  for (int i = 0; i < 4; ++i) xn[i] = x[i] * rstd * nw[d0 + i];
  bf16* dst = outh + ((size_t)hh * S + s) * D + d0;
#pragma unroll
  for (int i = 0; i < 4; ++i) {
    float other = __shfl_xor(xn[i], 16, 32);        // x[d ^ 64]
    float rot = (d0 < 64) ? -other : other;          // rotate_half
    float o = xn[i] * cosb[(size_t)s * D + d0 + i] + rot * sinb[(size_t)s * D + d0 + i];
    dst[i] = f2bf(o);
  }
}

// ---------------- V: bf16 + transpose to [hkv][d][s] ------------------------
__global__ void k_v_post(const float* __restrict__ v, bf16* __restrict__ vT) {
  size_t gid = (size_t)blockIdx.x * 256 + threadIdx.x;
  if (gid >= (size_t)S * KVD) return;
  int s  = (int)(gid / KVD);
  int cd = (int)(gid % KVD);             // kvh*D + d
  vT[(size_t)cd * S + s] = f2bf(v[gid]);
}

// ---------------- causal attention: 1 wave per (head, 16-row q block) -------
// 16 x 2048 fp32 score strip lives in LDS (128KB, fine on a 320KB WGP).
__global__ void k_attn(const bf16* __restrict__ qh, const bf16* __restrict__ kh,
                       const bf16* __restrict__ vT, bf16* __restrict__ ao) {
  extern __shared__ float sc[];          // [16][S]
  int h = blockIdx.x, qb = blockIdx.y;
  int q0 = qb * 16;
  int kvh = h / (HQ / HKV);
  int lane = threadIdx.x & 31;
  int col = lane & 15, hi = lane >> 4;
  v16bf qfr[4];
#pragma unroll
  for (int c = 0; c < 4; ++c)
    qfr[c] = load_a_frag(qh + ((size_t)h * S + q0) * D + c * 32, D);
  int nkt = qb + 1;
  const float scl = 0.08838834764831845f;  // 1/sqrt(128)
  for (int kt = 0; kt < nkt; ++kt) {
    v8f acc;
#pragma unroll
    for (int i = 0; i < 8; ++i) acc[i] = 0.f;
    const bf16* kb = kh + ((size_t)kvh * S + kt * 16) * D;
#pragma unroll
    for (int c = 0; c < 4; ++c)
      acc = wmma_bf16(qfr[c], load_bt_frag(kb + c * 32, D), acc);
#pragma unroll
    for (int j = 0; j < 8; ++j) {
      int r = j + hi * 8;
      int qi = q0 + r, kj = kt * 16 + col;
      float vv = acc[j] * scl;
      if (kj > qi) vv = -3.0e38f;
      sc[r * S + kt * 16 + col] = vv;
    }
  }
  __syncthreads();
  int kmax = nkt * 16;
  int r2 = lane >> 1, par = lane & 1;    // 2 lanes per row
  float mx = -3.4e38f;
  for (int c = par; c < kmax; c += 2) mx = fmaxf(mx, sc[r2 * S + c]);
  mx = fmaxf(mx, __shfl_xor(mx, 1, 32));
  float ssum = 0.f;
  for (int c = par; c < kmax; c += 2) { float e = __expf(sc[r2 * S + c] - mx); sc[r2 * S + c] = e; ssum += e; }
  ssum += __shfl_xor(ssum, 1, 32);
  float inv = 1.f / ssum;
  for (int c = par; c < kmax; c += 2) sc[r2 * S + c] *= inv;
  int kpad = (kmax + 31) & ~31;          // zero-pad odd 16-tiles up to K=32
  for (int c = kmax + par; c < kpad; c += 2) sc[r2 * S + c] = 0.f;
  __syncthreads();
  v8f o[8];
  for (int n = 0; n < 8; ++n)
    for (int i = 0; i < 8; ++i) o[n][i] = 0.f;
  int nkc = kpad >> 5;
  for (int kc = 0; kc < nkc; ++kc) {
    v16bf pa;
    int prow = lane & 15, pkb = (lane >> 4) * 8;
    const float* pb = &sc[prow * S + kc * 32];
#pragma unroll
    for (int i = 0; i < 8; ++i) { pa[i] = f2bf(pb[pkb + i]); pa[i + 8] = f2bf(pb[pkb + 16 + i]); }
#pragma unroll
    for (int n = 0; n < 8; ++n)
      o[n] = wmma_bf16(pa, load_bt_frag(vT + ((size_t)kvh * D + n * 16) * S + kc * 32, S), o[n]);
  }
#pragma unroll
  for (int n = 0; n < 8; ++n)
#pragma unroll
    for (int j = 0; j < 8; ++j) {
      int qi = q0 + j + hi * 8;
      ao[(size_t)qi * HQD + h * D + n * 16 + col] = f2bf(o[n][j]);
    }
}

// ---------------- residual add ----------------------------------------------
__global__ void k_add(const float* __restrict__ a, const float* __restrict__ b,
                      float* __restrict__ c, size_t n) {
  size_t g = (size_t)blockIdx.x * 256 + threadIdx.x;
  if (g < n) c[g] = a[g] + b[g];
}

// ---------------- gating: logits + softmax + top-4 --------------------------
__global__ void k_gate_topk(const bf16* __restrict__ h2, const float* __restrict__ gw,
                            int* __restrict__ ti, float* __restrict__ tw) {
  __shared__ float red[256];
  __shared__ float lg[E];
  int t = blockIdx.x, tid = threadIdx.x;
  const bf16* hr = h2 + (size_t)t * H;
  for (int e = 0; e < E; ++e) {
    float part = 0.f;
    for (int c = tid; c < H; c += 256) part += bf2f(hr[c]) * gw[(size_t)e * H + c];
    red[tid] = part; __syncthreads();
    for (int s2 = 128; s2 > 0; s2 >>= 1) { if (tid < s2) red[tid] += red[tid + s2]; __syncthreads(); }
    if (tid == 0) lg[e] = red[0];
    __syncthreads();
  }
  if (tid == 0) {
    float m = -3.4e38f;
    for (int e = 0; e < E; ++e) m = fmaxf(m, lg[e]);
    float pr[E]; float se = 0.f;
    for (int e = 0; e < E; ++e) { pr[e] = __expf(lg[e] - m); se += pr[e]; }
    float invs = 1.f / se;
    for (int e = 0; e < E; ++e) pr[e] *= invs;
    int idx[TK]; float w[TK]; float tot = 0.f;
    for (int k2 = 0; k2 < TK; ++k2) {
      int best = 0; float bv = -1.f;
      for (int e = 0; e < E; ++e) if (pr[e] > bv) { bv = pr[e]; best = e; }
      idx[k2] = best; w[k2] = bv; pr[best] = -2.f; tot += bv;
    }
    float invt = 1.f / tot;
    for (int k2 = 0; k2 < TK; ++k2) { ti[t * TK + k2] = idx[k2]; tw[t * TK + k2] = w[k2] * invt; }
  }
}

// ---------------- deterministic expert grouping (16-row padded) -------------
__global__ void k_group_build(const int* __restrict__ ti, const float* __restrict__ tw,
                              int* __restrict__ blk2exp, int* __restrict__ entry_pos,
                              int* __restrict__ rowtok, float* __restrict__ roww,
                              int* __restrict__ nblk) {
  __shared__ int cnt[E];
  __shared__ int off[E + 1];
  int e = threadIdx.x;  // 32 threads
  int c = 0;
  for (int p = 0; p < S * TK; ++p) if (ti[p] == e) c++;
  cnt[e] = c; __syncthreads();
  if (e == 0) {
    int cur = 0;
    for (int x2 = 0; x2 < E; ++x2) { off[x2] = cur; cur += (cnt[x2] + 15) & ~15; }
    off[E] = cur; *nblk = cur / 16;
    for (int x2 = 0; x2 < E; ++x2)
      for (int b = off[x2] / 16; b < off[x2 + 1] / 16; ++b) blk2exp[b] = x2;
  }
  __syncthreads();
  int pos = off[e];
  for (int p = 0; p < S * TK; ++p)
    if (ti[p] == e) { rowtok[pos] = p / TK; roww[pos] = tw[p]; entry_pos[p] = pos; pos++; }
  int end = off[e] + ((cnt[e] + 15) & ~15);
  for (; pos < end; ++pos) { rowtok[pos] = -1; roww[pos] = 0.f; }
}

// ---------------- MoE gate/up fused GEMM: gu = silu(x@Wg) * (x@Wu) ----------
__global__ void k_moe_gu(const bf16* __restrict__ h2, const bf16* __restrict__ wgT,
                         const bf16* __restrict__ wuT, const int* __restrict__ rowtok,
                         const int* __restrict__ blk2exp, const int* __restrict__ nblk,
                         bf16* __restrict__ gu) {
  int rb = blockIdx.y;
  if (rb >= *nblk) return;
  int nt = blockIdx.x;
  int e = blk2exp[rb];
  int lane = threadIdx.x & 31;
  int tok = rowtok[rb * 16 + (lane & 15)];
  const bf16* arow = (tok >= 0) ? (h2 + (size_t)tok * H + ((lane >> 4) * 8)) : (const bf16*)0;
  const bf16* Bg = wgT + (size_t)e * I * H + (size_t)nt * 16 * H;
  const bf16* Bu = wuT + (size_t)e * I * H + (size_t)nt * 16 * H;
  bf16 zb = f2bf(0.f);
  v8f ag, au;
#pragma unroll
  for (int i = 0; i < 8; ++i) { ag[i] = 0.f; au[i] = 0.f; }
  for (int kc = 0; kc < H / 32; ++kc) {
    v16bf af;
    if (tok >= 0) {
      v8bf lo = *(const v8bf*)(arow + kc * 32);
      v8bf hi2 = *(const v8bf*)(arow + kc * 32 + 16);
#pragma unroll
      for (int i = 0; i < 8; ++i) { af[i] = lo[i]; af[i + 8] = hi2[i]; }
    } else {
#pragma unroll
      for (int i = 0; i < 16; ++i) af[i] = zb;
    }
    ag = wmma_bf16(af, load_bt_frag(Bg + kc * 32, H), ag);   // EXEC uniform here
    au = wmma_bf16(af, load_bt_frag(Bu + kc * 32, H), au);
  }
  int col = lane & 15, hi = lane >> 4;
#pragma unroll
  for (int j = 0; j < 8; ++j) {
    float g = ag[j], u = au[j];
    float sg = g / (1.f + __expf(-g));
    gu[(size_t)(rb * 16 + j + hi * 8) * I + nt * 16 + col] = f2bf(sg * u);
  }
}

// ---------------- MoE down GEMM: dbuf = gu @ Wd -----------------------------
__global__ void k_moe_down(const bf16* __restrict__ gu, const bf16* __restrict__ wdT,
                           const int* __restrict__ blk2exp, const int* __restrict__ nblk,
                           float* __restrict__ dbuf) {
  int rb = blockIdx.y;
  if (rb >= *nblk) return;
  int nt = blockIdx.x;
  int e = blk2exp[rb];
  const bf16* Ab = gu + (size_t)rb * 16 * I;
  const bf16* Bb = wdT + (size_t)e * H * I + (size_t)nt * 16 * I;
  v8f acc;
#pragma unroll
  for (int i = 0; i < 8; ++i) acc[i] = 0.f;
  for (int kc = 0; kc < I / 32; ++kc)
    acc = wmma_bf16(load_a_frag(Ab + kc * 32, I), load_bt_frag(Bb + kc * 32, I), acc);
  int lane = threadIdx.x & 31, col = lane & 15, hi = lane >> 4;
#pragma unroll
  for (int j = 0; j < 8; ++j)
    dbuf[(size_t)(rb * 16 + j + hi * 8) * H + nt * 16 + col] = acc[j];
}

// ---------------- deterministic gather-combine (replaces scatter-add) -------
__global__ void k_moe_combine(const float* __restrict__ x1, const float* __restrict__ dbuf,
                              const int* __restrict__ entry_pos, const float* __restrict__ roww,
                              float* __restrict__ out) {
  size_t gid = (size_t)blockIdx.x * 256 + threadIdx.x;
  if (gid >= (size_t)S * H) return;
  int t = (int)(gid / H), c = (int)(gid % H);
  float a = x1[gid];
#pragma unroll
  for (int k2 = 0; k2 < TK; ++k2) {
    int p = entry_pos[t * TK + k2];
    a += roww[p] * dbuf[(size_t)p * H + c];
  }
  out[gid] = a;
}

// ---------------------------------------------------------------------------
extern "C" void kernel_launch(void* const* d_in, const int* in_sizes, int n_in,
                              void* d_out, int out_size, void* d_ws, size_t ws_size,
                              hipStream_t stream) {
  (void)in_sizes; (void)n_in; (void)out_size; (void)ws_size;
  const float* hs      = (const float*)d_in[0];
  const float* cosb    = (const float*)d_in[1];
  const float* sinb    = (const float*)d_in[2];
  const float* in_ln_w = (const float*)d_in[3];
  const float* post_ln = (const float*)d_in[4];
  const float* qn_w    = (const float*)d_in[5];
  const float* kn_w    = (const float*)d_in[6];
  const float* wq      = (const float*)d_in[7];
  const float* wk      = (const float*)d_in[8];
  const float* wv      = (const float*)d_in[9];
  const float* wo      = (const float*)d_in[10];
  const float* gate_w  = (const float*)d_in[11];
  const float* wg      = (const float*)d_in[12];
  const float* wu      = (const float*)d_in[13];
  const float* wd      = (const float*)d_in[14];
  float* out = (float*)d_out;

  // workspace carve
  size_t cur = 0;
  auto carve = [&](size_t bytes) {
    void* p = (char*)d_ws + cur;
    cur = (cur + bytes + 255) & ~(size_t)255;
    return p;
  };
  bf16* wqT = (bf16*)carve((size_t)HQD * H * 2);
  bf16* wkT = (bf16*)carve((size_t)KVD * H * 2);
  bf16* wvT = (bf16*)carve((size_t)KVD * H * 2);
  bf16* woT = (bf16*)carve((size_t)H * HQD * 2);
  bf16* wgT = (bf16*)carve((size_t)E * I * H * 2);
  bf16* wuT = (bf16*)carve((size_t)E * I * H * 2);
  bf16* wdT = (bf16*)carve((size_t)E * H * I * 2);
  bf16* h_bf  = (bf16*)carve((size_t)S * H * 2);
  float* q_f  = (float*)carve((size_t)S * HQD * 4);
  float* k_f  = (float*)carve((size_t)S * KVD * 4);
  float* v_f  = (float*)carve((size_t)S * KVD * 4);
  bf16* qh    = (bf16*)carve((size_t)HQ * S * D * 2);
  bf16* kh    = (bf16*)carve((size_t)HKV * S * D * 2);
  bf16* vT    = (bf16*)carve((size_t)HKV * D * S * 2);
  bf16* ao    = (bf16*)carve((size_t)S * HQD * 2);
  float* attn_f = (float*)carve((size_t)S * H * 4);
  float* x1_f   = (float*)carve((size_t)S * H * 4);
  bf16* h2_bf   = (bf16*)carve((size_t)S * H * 2);
  int*   ti = (int*)carve((size_t)S * TK * 4);
  float* tw = (float*)carve((size_t)S * TK * 4);
  int* blk2exp   = (int*)carve((size_t)MAXBLK * 4);
  int* entry_pos = (int*)carve((size_t)S * TK * 4);
  int* rowtok    = (int*)carve((size_t)PADROWS * 4);
  float* roww    = (float*)carve((size_t)PADROWS * 4);
  int* nblk      = (int*)carve(256);
  bf16* gu_bf    = (bf16*)carve((size_t)PADROWS * I * 2);
  float* dbuf    = (float*)carve((size_t)PADROWS * H * 4);

  // 1) weight convert/transpose to bf16 B^T layouts
  k_transpose_bf16<<<dim3((HQD * H) / 256, 1, 1), 256, 0, stream>>>(wq, wqT, H, HQD);
  k_transpose_bf16<<<dim3((KVD * H) / 256, 1, 1), 256, 0, stream>>>(wk, wkT, H, KVD);
  k_transpose_bf16<<<dim3((KVD * H) / 256, 1, 1), 256, 0, stream>>>(wv, wvT, H, KVD);
  k_transpose_bf16<<<dim3((HQD * H) / 256, 1, 1), 256, 0, stream>>>(wo, woT, HQD, H);
  k_transpose_bf16<<<dim3((I * H) / 256, 1, E), 256, 0, stream>>>(wg, wgT, H, I);
  k_transpose_bf16<<<dim3((I * H) / 256, 1, E), 256, 0, stream>>>(wu, wuT, H, I);
  k_transpose_bf16<<<dim3((I * H) / 256, 1, E), 256, 0, stream>>>(wd, wdT, I, H);

  // 2) input RMSNorm -> bf16, then QKV WMMA GEMMs (32x64 per wave, async LDS A)
  k_rmsnorm_bf16<<<S, 256, 0, stream>>>(hs, in_ln_w, h_bf, H);
  k_gemm_bt<<<dim3(HQD / 64, S / 32), 32, 0, stream>>>(h_bf, wqT, q_f, HQD, H);
  k_gemm_bt<<<dim3(KVD / 64, S / 32), 32, 0, stream>>>(h_bf, wkT, k_f, KVD, H);
  k_gemm_bt<<<dim3(KVD / 64, S / 32), 32, 0, stream>>>(h_bf, wvT, v_f, KVD, H);

  // 3) per-head RMSNorm + RoPE (wave32 shuffles), V transpose
  k_qk_post<<<(S * HQ) / 4, 128, 0, stream>>>(q_f, qn_w, cosb, sinb, qh, HQ);
  k_qk_post<<<(S * HKV) / 4, 128, 0, stream>>>(k_f, kn_w, cosb, sinb, kh, HKV);
  k_v_post<<<(S * KVD) / 256, 256, 0, stream>>>(v_f, vT);

  // 4) causal attention (128KB LDS score strip per block)
  k_attn<<<dim3(HQ, S / 16), 32, 16 * S * sizeof(float), stream>>>(qh, kh, vT, ao);

  // 5) O projection + residual
  k_gemm_bt<<<dim3(H / 64, S / 32), 32, 0, stream>>>(ao, woT, attn_f, H, HQD);
  k_add<<<(S * H) / 256, 256, 0, stream>>>(hs, attn_f, x1_f, (size_t)S * H);

  // 6) MoE: post-norm, gate/topk, deterministic grouping, expert GEMMs, combine
  k_rmsnorm_bf16<<<S, 256, 0, stream>>>(x1_f, post_ln, h2_bf, H);
  k_gate_topk<<<S, 256, 0, stream>>>(h2_bf, gate_w, ti, tw);
  k_group_build<<<1, 32, 0, stream>>>(ti, tw, blk2exp, entry_pos, rowtok, roww, nblk);
  k_moe_gu<<<dim3(I / 16, MAXBLK), 32, 0, stream>>>(h2_bf, wgT, wuT, rowtok, blk2exp, nblk, gu_bf);
  k_moe_down<<<dim3(H / 16, MAXBLK), 32, 0, stream>>>(gu_bf, wdT, blk2exp, nblk, dbuf);
  k_moe_combine<<<(S * H) / 256, 256, 0, stream>>>(x1_f, dbuf, entry_pos, roww, out);
}
